// MFN_18829136626419
// MI455X (gfx1250) — matrix operand order repaired and checked by reading, not compiled, not run on hardware
//
#include <hip/hip_runtime.h>

// ---------------------------------------------------------------------------
// MFN (3x LSTM + attention/gating MLPs) recurrent network on gfx1250.
//  - batch tiled 16 rows per workgroup (WMMA M=16), 128 WGs, 8 waves each
//  - all recurrent state (h, c, mem) kept in LDS for the full T=100 loop
//  - all GEMMs: v_wmma_f32_16x16x32_bf16 (bf16 in, f32 accumulate)
//  - weights pre-swizzled once into WMMA B-fragment layout in d_ws (L2-resident)
//  - parallel softmax (half-wave per row, shfl_xor reductions)
//  - next-step x tile prefetched (global_prefetch) during compute
// ---------------------------------------------------------------------------

typedef __bf16 bf16_t;
typedef __attribute__((ext_vector_type(16))) __bf16 v16bf;
typedef __attribute__((ext_vector_type(8)))  __bf16 v8bf;
typedef __attribute__((ext_vector_type(8)))  float  v8f;

#define T_STEPS 100
#define BATCH   2048
#define DLX     300
#define DAX     74
#define DVX     35
#define DX      409       // D_L + D_A + D_V
#define DHL     128
#define DHA     32
#define DHV     32
#define TOTALH  192       // DHL+DHA+DHV
#define MEMD    128
#define ATTIN   384       // 2*TOTALH
#define GAMIN   512       // ATTIN + MEMD
#define FINALD  320       // TOTALH + MEMD
#define HNN     256

#define NW      8         // waves per block (wave32)
#define NTHR    256

// x staging buffer layout: [x_l pad->320 | x_a pad->96 | x_v pad->64]
#define XS      480
#define XO_A    320
#define XO_V    416

struct WTab {
  const unsigned short *wih_l, *whh_l, *wih_a, *whh_a, *wih_v, *whh_v;
  const unsigned short *a1w1, *a1w2, *a2w1, *a2w2;
  const unsigned short *g1w1, *g1w2, *g2w1, *g2w2, *ow1, *ow2;
};
struct BTab {
  const float *bih_l, *bhh_l, *bih_a, *bhh_a, *bih_v, *bhh_v;
  const float *a1b1, *a1b2, *a2b1, *a2b2;
  const float *g1b1, *g1b2, *g2b1, *g2b2, *ob1, *ob2;
};

__device__ __forceinline__ float sigf(float x)   { return 1.0f / (1.0f + __expf(-x)); }
__device__ __forceinline__ float tanhff(float x) { return 1.0f - 2.0f / (__expf(2.0f * x) + 1.0f); }

// ---------------------------------------------------------------------------
// One (16 x 16*NT) GEMM tile set:  Out = A1(16xK1) * W1^T  [+ A2(16xK2) * W2^T]
// A buffers: row-major bf16 in LDS.  W buffers: pre-swizzled B-fragments in
// global memory (per (nt,kt) block: 32 lanes x 16 halves, 32B/lane contiguous).
// C written to f32 LDS panel (stride os, column offset ocol).
// ---------------------------------------------------------------------------
template<int NK1, int NK2>
__device__ __forceinline__ void gemm_t(const bf16_t* __restrict__ A1, int as1,
                                       const unsigned short* __restrict__ W1u,
                                       const bf16_t* __restrict__ A2, int as2,
                                       const unsigned short* __restrict__ W2u,
                                       float* __restrict__ outF, int os, int ocol,
                                       int NT, int tid)
{
  const int lane = tid & 31;
  const int wave = tid >> 5;
  const int row  = lane & 15;   // C: lane -> column; A: lane -> row (M)
  const int sel  = lane >> 4;   // lane-group select for K halves
  const bf16_t* W1 = (const bf16_t*)W1u;
  const bf16_t* W2 = (const bf16_t*)W2u;

  for (int nt = wave; nt < NT; nt += NW) {
    v8f acc = {};
#pragma unroll
    for (int kt = 0; kt < NK1; ++kt) {
      const v8bf* p0 = (const v8bf*)(A1 + row * as1 + kt * 32 + sel * 16);
      const v8bf* p1 = (const v8bf*)(A1 + row * as1 + kt * 32 + sel * 16 + 8);
      v16bf a = __builtin_shufflevector(*p0, *p1, 0,1,2,3,4,5,6,7,8,9,10,11,12,13,14,15);
      v16bf b = *(const v16bf*)(W1 + ((size_t)(nt * NK1 + kt) * 32 + lane) * 16);
      acc = __builtin_amdgcn_wmma_f32_16x16x32_bf16(false, a, false, b, (short)0, acc, false, false);
    }
    if constexpr (NK2 > 0) {
#pragma unroll
      for (int kt = 0; kt < NK2; ++kt) {
        const v8bf* p0 = (const v8bf*)(A2 + row * as2 + kt * 32 + sel * 16);
        const v8bf* p1 = (const v8bf*)(A2 + row * as2 + kt * 32 + sel * 16 + 8);
        v16bf a = __builtin_shufflevector(*p0, *p1, 0,1,2,3,4,5,6,7,8,9,10,11,12,13,14,15);
        v16bf b = *(const v16bf*)(W2 + ((size_t)(nt * NK2 + kt) * 32 + lane) * 16);
        acc = __builtin_amdgcn_wmma_f32_16x16x32_bf16(false, a, false, b, (short)0, acc, false, false);
      }
    }
    const int n = ocol + nt * 16 + row;   // C layout: lane 0-15 N=lane, 16-31 N=lane-16
#pragma unroll
    for (int v = 0; v < 8; ++v)
      outF[(sel * 8 + v) * os + n] = acc[v];   // M = sel*8 + v
  }
}

// ---------------------------------------------------------------------------
// Weight pre-swizzle: f32 row-major W (N x K) -> bf16 WMMA B-fragment layout.
//  frag[(nt*nkt+kt)*512 + lane*16 + j] = W[nt*16 + (lane&15), kt*32 + (lane>>4)*16 + j]
// Zero-pads K to 32 and N to 16.
// ---------------------------------------------------------------------------
__global__ void prep_weight(const float* __restrict__ W, unsigned short* __restrict__ dst,
                            int N, int K, int nkt, int ntt)
{
  const int total = ntt * nkt * 512;
  for (int idx = blockIdx.x * blockDim.x + threadIdx.x; idx < total;
       idx += gridDim.x * blockDim.x) {
    int j    = idx & 15;
    int lane = (idx >> 4) & 31;
    int blk  = idx >> 9;
    int kt   = blk % nkt;
    int nt   = blk / nkt;
    int n = nt * 16 + (lane & 15);
    int k = kt * 32 + (lane >> 4) * 16 + j;
    float v = (n < N && k < K) ? W[n * K + k] : 0.0f;
    bf16_t h = (bf16_t)v;
    dst[idx] = __builtin_bit_cast(unsigned short, h);
  }
}

// ---------------------------------------------------------------------------
// Main recurrent kernel: one block = 16 batch rows, runs all 100 steps.
// ---------------------------------------------------------------------------
__global__ __launch_bounds__(NTHR, 1)
void mfn_kernel(const float* __restrict__ x, float* __restrict__ out, WTab w, BTab b)
{
  // bf16 A-operand staging
  __shared__ __align__(16) bf16_t xH[16 * XS];         // x tile, padded segments
  __shared__ __align__(16) bf16_t hH[16 * TOTALH];     // [h_l | h_a | h_v]
  __shared__ __align__(16) bf16_t cStarH[16 * ATTIN];
  __shared__ __align__(16) bf16_t bothH[16 * GAMIN];   // [attended(384) | mem(128)]
  __shared__ __align__(16) bf16_t hidH[16 * HNN];
  // f32 panels / state
  __shared__ float F1[16 * 512];                       // GEMM out A (gates / hidden)
  __shared__ float F2[16 * 384];                       // GEMM out B (scores / heads)
  __shared__ float cF[16 * 320];                       // [c_l | c_a | c_v | mem]
  __shared__ float cNw[16 * 256];                      // nc (0..191); later cHat|gamma1

  const int tid  = threadIdx.x;
  const int b0   = blockIdx.x * 16;
  const int lane = tid & 31;
  const int wave = tid >> 5;

  // zero-init state
  for (int i = tid; i < 16 * 320; i += NTHR) cF[i] = 0.0f;
  for (int i = tid; i < 16 * TOTALH; i += NTHR) hH[i] = (bf16_t)0.0f;
  for (int i = tid; i < 16 * MEMD; i += NTHR) {
    int r = i >> 7, c = i & 127;
    bothH[r * GAMIN + ATTIN + c] = (bf16_t)0.0f;
  }
  __syncthreads();

  for (int t = 0; t < T_STEPS; ++t) {
    // ---- stage x_t tile (f32 HBM/L2 -> bf16 LDS, zero-padded K segments) ----
    const float* xt = x + ((size_t)t * BATCH + b0) * DX;
    for (int i = tid; i < 16 * XS; i += NTHR) {
      int r = i / XS, c = i % XS;
      float v = 0.0f;
      if (c < DLX)                          v = xt[r * DX + c];
      else if (c >= XO_A && c < XO_A + DAX) v = xt[r * DX + DLX + (c - XO_A)];
      else if (c >= XO_V && c < XO_V + DVX) v = xt[r * DX + DLX + DAX + (c - XO_V)];
      xH[i] = (bf16_t)v;
    }
    // prefetch next step's x tile into cache while this step computes
    if (t + 1 < T_STEPS) {
      const char* nx = (const char*)(x + ((size_t)(t + 1) * BATCH + b0) * DX);
      for (int off = tid * 256; off < 16 * DX * 4; off += NTHR * 256)
        __builtin_prefetch(nx + off, 0, 1);
    }
    __syncthreads();

    // ---- LSTM gate GEMMs: g = x@Wih^T + h@Whh^T (biases added in act pass) ----
    gemm_t<10, 4>(xH,         XS, w.wih_l, hH,        TOTALH, w.whh_l, F1, 512, 0,   32, tid);
    gemm_t<3,  1>(xH + XO_A,  XS, w.wih_a, hH + 128,  TOTALH, w.whh_a, F2, 384, 0,    8, tid);
    gemm_t<2,  1>(xH + XO_V,  XS, w.wih_v, hH + 160,  TOTALH, w.whh_v, F2, 384, 128,  8, tid);
    __syncthreads();

    // ---- LSTM activations (i,f,g,o), nc -> cNw, nh -> hH (bf16) ----
    for (int i = tid; i < 16 * TOTALH; i += NTHR) {
      int r = i / TOTALH, c = i % TOTALH;
      const float *gb, *bi, *bh; int dh, ci;
      if (c < 128)      { gb = F1 + r * 512;       bi = b.bih_l; bh = b.bhh_l; dh = 128; ci = c;       }
      else if (c < 160) { gb = F2 + r * 384;       bi = b.bih_a; bh = b.bhh_a; dh = 32;  ci = c - 128; }
      else              { gb = F2 + r * 384 + 128; bi = b.bih_v; bh = b.bhh_v; dh = 32;  ci = c - 160; }
      float iv = sigf  (gb[ci]          + bi[ci]          + bh[ci]);
      float fv = sigf  (gb[dh + ci]     + bi[dh + ci]     + bh[dh + ci]);
      float gv = tanhff(gb[2 * dh + ci] + bi[2 * dh + ci] + bh[2 * dh + ci]);
      float ov = sigf  (gb[3 * dh + ci] + bi[3 * dh + ci] + bh[3 * dh + ci]);
      float nc = fv * cF[r * 320 + c] + iv * gv;
      cNw[r * 256 + c]   = nc;
      hH[r * TOTALH + c] = (bf16_t)(ov * tanhff(nc));
    }
    __syncthreads();

    // ---- cStar = [c_old(192) | c_new(192)] (bf16 A-buffer) ----
    for (int i = tid; i < 16 * ATTIN; i += NTHR) {
      int r = i / ATTIN, c = i % ATTIN;
      float v = (c < TOTALH) ? cF[r * 320 + c] : cNw[r * 256 + (c - TOTALH)];
      cStarH[i] = (bf16_t)v;
    }
    __syncthreads();
    // commit c <- nc (cNw becomes free scratch after this)
    for (int i = tid; i < 16 * TOTALH; i += NTHR) {
      int r = i / TOTALH, c = i % TOTALH;
      cF[r * 320 + c] = cNw[r * 256 + c];
    }
    __syncthreads();

    // ---- attention MLP: hid = relu(cStar@a1w1^T + b) ; scores = hid@a1w2^T ----
    gemm_t<12, 0>(cStarH, ATTIN, w.a1w1, nullptr, 0, nullptr, F1, 512, 0, 16, tid);
    __syncthreads();
    for (int i = tid; i < 16 * HNN; i += NTHR) {
      int r = i / HNN, c = i % HNN;
      float v = F1[r * 512 + c] + b.a1b1[c];
      hidH[i] = (bf16_t)(v > 0.0f ? v : 0.0f);
    }
    __syncthreads();
    gemm_t<8, 0>(hidH, HNN, w.a1w2, nullptr, 0, nullptr, F2, 384, 0, 24, tid);
    __syncthreads();

    // ---- parallel softmax: half-wave (16 lanes) per row, shfl_xor reduce ----
    {
      const int r   = wave * 2 + (lane >> 4);  // 8 waves x 2 rows = 16 rows
      const int sub = lane & 15;
      float mx = -3.4e38f;
      for (int c = sub; c < ATTIN; c += 16) {
        float v = F2[r * 384 + c] + b.a1b2[c];
        mx = v > mx ? v : mx;
      }
#pragma unroll
      for (int off = 8; off > 0; off >>= 1) {
        float o = __shfl_xor(mx, off, 16);
        mx = o > mx ? o : mx;
      }
      float s = 0.0f;
      for (int c = sub; c < ATTIN; c += 16)
        s += __expf(F2[r * 384 + c] + b.a1b2[c] - mx);
#pragma unroll
      for (int off = 8; off > 0; off >>= 1)
        s += __shfl_xor(s, off, 16);
      float inv = 1.0f / s;
      for (int c = sub; c < ATTIN; c += 16) {
        float att = __expf(F2[r * 384 + c] + b.a1b2[c] - mx) * inv;
        bothH[r * GAMIN + c] = (bf16_t)(att * (float)cStarH[r * ATTIN + c]);
      }
    }
    __syncthreads();

    // ---- cHat = tanh(mlp(attended)) ----
    gemm_t<12, 0>(bothH, GAMIN, w.a2w1, nullptr, 0, nullptr, F1, 512, 0, 16, tid);
    __syncthreads();
    for (int i = tid; i < 16 * HNN; i += NTHR) {
      int r = i / HNN, c = i % HNN;
      float v = F1[r * 512 + c] + b.a2b1[c];
      hidH[i] = (bf16_t)(v > 0.0f ? v : 0.0f);
    }
    __syncthreads();
    gemm_t<8, 0>(hidH, HNN, w.a2w2, nullptr, 0, nullptr, F2, 384, 0, 8, tid);
    __syncthreads();
    // cHat -> cNw[0:128]; overlap with g1 hidden GEMM (independent buffers)
    for (int i = tid; i < 16 * MEMD; i += NTHR) {
      int r = i >> 7, c = i & 127;
      cNw[r * 256 + c] = tanhff(F2[r * 384 + c] + b.a2b2[c]);
    }
    gemm_t<16, 0>(bothH, GAMIN, w.g1w1, nullptr, 0, nullptr, F1, 512, 0, 16, tid);
    __syncthreads();
    for (int i = tid; i < 16 * HNN; i += NTHR) {
      int r = i / HNN, c = i % HNN;
      float v = F1[r * 512 + c] + b.g1b1[c];
      hidH[i] = (bf16_t)(v > 0.0f ? v : 0.0f);
    }
    __syncthreads();
    gemm_t<8, 0>(hidH, HNN, w.g1w2, nullptr, 0, nullptr, F2, 384, 0, 8, tid);
    __syncthreads();
    // gamma1 -> cNw[128:256]; overlap with g2 hidden GEMM
    for (int i = tid; i < 16 * MEMD; i += NTHR) {
      int r = i >> 7, c = i & 127;
      cNw[r * 256 + 128 + c] = sigf(F2[r * 384 + c] + b.g1b2[c]);
    }
    gemm_t<16, 0>(bothH, GAMIN, w.g2w1, nullptr, 0, nullptr, F1, 512, 0, 16, tid);
    __syncthreads();
    for (int i = tid; i < 16 * HNN; i += NTHR) {
      int r = i / HNN, c = i % HNN;
      float v = F1[r * 512 + c] + b.g2b1[c];
      hidH[i] = (bf16_t)(v > 0.0f ? v : 0.0f);
    }
    __syncthreads();
    gemm_t<8, 0>(hidH, HNN, w.g2w2, nullptr, 0, nullptr, F2, 384, 0, 8, tid);
    __syncthreads();

    // ---- mem = gamma1*mem + gamma2*cHat ; refresh mem bf16 in bothH[384:512] ----
    for (int i = tid; i < 16 * MEMD; i += NTHR) {
      int r = i >> 7, c = i & 127;
      float g2v = sigf(F2[r * 384 + c] + b.g2b2[c]);
      float m = cNw[r * 256 + 128 + c] * cF[r * 320 + TOTALH + c] + g2v * cNw[r * 256 + c];
      cF[r * 320 + TOTALH + c]     = m;
      bothH[r * GAMIN + ATTIN + c] = (bf16_t)m;
    }
    __syncthreads();
  }

  // ---- output MLP: last_hs = [h_l, h_a, h_v, mem] (320) ----
  for (int i = tid; i < 16 * FINALD; i += NTHR) {
    int r = i / FINALD, c = i % FINALD;
    float v = (c < TOTALH) ? (float)hH[r * TOTALH + c] : cF[r * 320 + c];
    xH[r * XS + c] = (bf16_t)v;
  }
  __syncthreads();
  gemm_t<10, 0>(xH, XS, w.ow1, nullptr, 0, nullptr, F1, 512, 0, 16, tid);
  __syncthreads();
  for (int i = tid; i < 16 * HNN; i += NTHR) {
    int r = i / HNN, c = i % HNN;
    float v = F1[r * 512 + c] + b.ob1[c];
    hidH[i] = (bf16_t)(v > 0.0f ? v : 0.0f);
  }
  __syncthreads();
  gemm_t<8, 0>(hidH, HNN, w.ow2, nullptr, 0, nullptr, F2, 384, 0, 1, tid);
  __syncthreads();
  if (tid < 16) out[b0 + tid] = F2[tid * 384 + 0] + b.ob2[0];
}

// ---------------------------------------------------------------------------
extern "C" void kernel_launch(void* const* d_in, const int* in_sizes, int n_in,
                              void* d_out, int out_size, void* d_ws, size_t ws_size,
                              hipStream_t stream)
{
  (void)in_sizes; (void)n_in; (void)out_size; (void)ws_size;
  const float* x = (const float*)d_in[0];

  unsigned short* ws = (unsigned short*)d_ws;
  size_t off = 0;
  auto prep = [&](const float* W, int N, int K) -> const unsigned short* {
    int nkt = (K + 31) / 32;
    int ntt = (N + 15) / 16;
    unsigned short* dst = ws + off;
    size_t total = (size_t)ntt * nkt * 512;
    off += total;
    int blocks = (int)((total + 255) / 256);
    if (blocks > 1024) blocks = 1024;
    prep_weight<<<blocks, 256, 0, stream>>>(W, dst, N, K, nkt, ntt);
    return dst;
  };

  WTab w;
  w.wih_l = prep((const float*)d_in[1],  4 * DHL, DLX);
  w.whh_l = prep((const float*)d_in[3],  4 * DHL, DHL);
  w.wih_a = prep((const float*)d_in[5],  4 * DHA, DAX);
  w.whh_a = prep((const float*)d_in[7],  4 * DHA, DHA);
  w.wih_v = prep((const float*)d_in[9],  4 * DHV, DVX);
  w.whh_v = prep((const float*)d_in[11], 4 * DHV, DHV);
  w.a1w1  = prep((const float*)d_in[13], HNN,   ATTIN);
  w.a1w2  = prep((const float*)d_in[15], ATTIN, HNN);
  w.a2w1  = prep((const float*)d_in[17], HNN,   ATTIN);
  w.a2w2  = prep((const float*)d_in[19], MEMD,  HNN);
  w.g1w1  = prep((const float*)d_in[21], HNN,   GAMIN);
  w.g1w2  = prep((const float*)d_in[23], MEMD,  HNN);
  w.g2w1  = prep((const float*)d_in[25], HNN,   GAMIN);
  w.g2w2  = prep((const float*)d_in[27], MEMD,  HNN);
  w.ow1   = prep((const float*)d_in[29], HNN,   FINALD);
  w.ow2   = prep((const float*)d_in[31], 1,     HNN);

  BTab b;
  b.bih_l = (const float*)d_in[2];   b.bhh_l = (const float*)d_in[4];
  b.bih_a = (const float*)d_in[6];   b.bhh_a = (const float*)d_in[8];
  b.bih_v = (const float*)d_in[10];  b.bhh_v = (const float*)d_in[12];
  b.a1b1  = (const float*)d_in[14];  b.a1b2  = (const float*)d_in[16];
  b.a2b1  = (const float*)d_in[18];  b.a2b2  = (const float*)d_in[20];
  b.g1b1  = (const float*)d_in[22];  b.g1b2  = (const float*)d_in[24];
  b.g2b1  = (const float*)d_in[26];  b.g2b2  = (const float*)d_in[28];
  b.ob1   = (const float*)d_in[30];  b.ob2   = (const float*)d_in[32];

  mfn_kernel<<<BATCH / 16, NTHR, 0, stream>>>(x, (float*)d_out, w, b);
}
